// GrassmannianLayerSelector_12283606467948
// MI455X (gfx1250) — compile-verified, matrix operand order checked by reading, not run on hardware
//
#include <hip/hip_runtime.h>
#include <math.h>

// ---------------- types ----------------
typedef float    v2f  __attribute__((ext_vector_type(2)));
typedef float    v4f  __attribute__((ext_vector_type(4)));
typedef float    v8f  __attribute__((ext_vector_type(8)));
typedef _Float16 v16h __attribute__((ext_vector_type(16)));

// ---------------- problem constants ----------------
#define LN    12            // teacher layers
#define PN    4             // student points
#define BN    32
#define NTOK  197
#define HN    12
#define DT    768
#define DS    384
#define MM    6304          // B*N
#define PD    128           // PROJ_DIM
#define KR    16            // SUBSPACE_RANK
#define COVEPS 1e-4f

#define MTX   ((size_t)BN*NTOK*DT)        // 4,841,472  per-layer teacher token elems
#define MAX_  ((size_t)BN*HN*NTOK*NTOK)   // 14,902,656 per-layer attn elems

// ---------------- WMMA helpers ----------------
static __device__ __forceinline__ v8f wmma_f32_k4(v2f a, v2f b, v8f c) {
  // D(16x16,f32) += A(16x4,f32) x B(4x16,f32)
  return __builtin_amdgcn_wmma_f32_16x16x4_f32(false, a, false, b, (short)0, c, false, false);
}
static __device__ __forceinline__ v8f wmma_f16_k32(v16h a, v16h b, v8f c) {
  return __builtin_amdgcn_wmma_f32_16x16x32_f16(false, a, false, b, (short)0, c, false, false);
}
// 16-bit A/B fragment K-index map (ISA 7.12.2, 16-bit A 16x32)
static __device__ __forceinline__ int kmap16(int j, bool hi) {
  return hi ? ((j < 8) ? j + 8 : j + 16) : ((j < 8) ? j : j + 8);
}

// =====================================================================
// K1/K2: z[mat][m][q] = sum_d tok[mat][m][d] * phi[q][d]   (WMMA f32 16x16x4)
// grid = (MM/16, nmat), block = 256 (8 waves, one 16x16 q-tile per wave)
// =====================================================================
#define APAD 772
__global__ void proj_kernel(const float* __restrict__ tok,
                            const float* __restrict__ phi,
                            float* __restrict__ zout, int D) {
  __shared__ float At[16 * APAD];
  const int tid = threadIdx.x;
  const int mat = blockIdx.y;
  const int m0  = blockIdx.x * 16;

  for (int idx = tid; idx < 16 * D; idx += 256) {
    int r = idx / D, c = idx - r * D;
    At[r * APAD + c] = tok[((size_t)mat * MM + m0 + r) * (size_t)D + c];
  }
  __syncthreads();

  const int  wave = tid >> 5, lane = tid & 31;
  const bool hi   = lane >= 16;
  const int  nn   = lane & 15;
  const int  q0   = wave * 16;
  const int  koff = hi ? 2 : 0;

  v8f acc = {0.f, 0.f, 0.f, 0.f, 0.f, 0.f, 0.f, 0.f};
  const float* aRow   = At + nn * APAD;            // A row m = lane%16
  const float* phiRow = phi + (size_t)(q0 + nn) * D; // B col n = lane%16

  for (int kk = 0; kk < D; kk += 4) {
    v2f a, b;
    a[0] = aRow[kk + koff];     a[1] = aRow[kk + koff + 1];
    b[0] = phiRow[kk + koff];   b[1] = phiRow[kk + koff + 1];
    acc = wmma_f32_k4(a, b, acc);
  }
  const int rbase = hi ? 8 : 0;
  for (int v = 0; v < 8; ++v) {
    int row = v + rbase;
    zout[((size_t)mat * MM + m0 + row) * PD + q0 + nn] = acc[v];
  }
}

// =====================================================================
// K3: per-matrix mean + covariance: cov = Z^T Z / M - mean mean^T + eps I
// grid = 16 matrices, block = 256 (8 waves; wave w owns q-tile w, loops 8 r-tiles)
// =====================================================================
__global__ void cov_kernel(const float* __restrict__ z,
                           float* __restrict__ cov,
                           float* __restrict__ meanOut) {
  __shared__ float Zt[64 * 128];
  __shared__ float meanL[128];
  const int tid = threadIdx.x;
  const int mat = blockIdx.x;
  const int wave = tid >> 5, lane = tid & 31;
  const bool hi = lane >= 16;
  const int nn = lane & 15;
  const int koff = hi ? 2 : 0;
  const float* zm = z + (size_t)mat * MM * PD;

  v8f acc[8];
  for (int r = 0; r < 8; ++r) acc[r] = (v8f){0.f,0.f,0.f,0.f,0.f,0.f,0.f,0.f};
  float csum = 0.f;

  const int nchunk = (MM + 63) / 64;  // 99, last padded with zeros
  for (int ch = 0; ch < nchunk; ++ch) {
    int m0 = ch * 64;
    __syncthreads();
    for (int idx = tid; idx < 64 * 128; idx += 256) {
      int r = idx >> 7, c = idx & 127;
      int mg = m0 + r;
      Zt[idx] = (mg < MM) ? zm[(size_t)mg * PD + c] : 0.f;
    }
    __syncthreads();
    if (tid < 128) {
      float s = 0.f;
      for (int r = 0; r < 64; ++r) s += Zt[r * 128 + tid];
      csum += s;
    }
    for (int k4 = 0; k4 < 64; k4 += 4) {
      int kk = k4 + koff;
      v2f a;
      a[0] = Zt[kk * 128 + wave * 16 + nn];
      a[1] = Zt[(kk + 1) * 128 + wave * 16 + nn];
      for (int rt = 0; rt < 8; ++rt) {
        v2f b;
        b[0] = Zt[kk * 128 + rt * 16 + nn];
        b[1] = Zt[(kk + 1) * 128 + rt * 16 + nn];
        acc[rt] = wmma_f32_k4(a, b, acc[rt]);
      }
    }
  }
  if (tid < 128) {
    float mv = csum / (float)MM;
    meanL[tid] = mv;
    meanOut[mat * 128 + tid] = mv;
  }
  __syncthreads();
  const float invM = 1.f / (float)MM;
  const int rbase = hi ? 8 : 0;
  for (int rt = 0; rt < 8; ++rt)
    for (int v = 0; v < 8; ++v) {
      int qq = wave * 16 + v + rbase;
      int rr = rt * 16 + nn;
      float val = acc[rt][v] * invM - meanL[qq] * meanL[rr] + ((qq == rr) ? COVEPS : 0.f);
      cov[(size_t)mat * (128 * 128) + qq * 128 + rr] = val;
    }
}

// =====================================================================
// K4: parallel two-sided Jacobi eigensolver (128x128) + top-16 column select
// grid = 16 matrices, block = 512, dynamic LDS = 2*128*129*4 bytes
// =====================================================================
static __device__ __forceinline__ void pq_of(int k, int round, int* p, int* q) {
  *p = (k == 0) ? 0 : 1 + (k - 1 + round) % 127;
  *q = 1 + (126 - k + round) % 127;
}

__global__ void jacobi_kernel(const float* __restrict__ cov, float* __restrict__ U) {
  extern __shared__ float lds[];
  float* Am = lds;                 // 128 x 129
  float* Vm = lds + 128 * 129;     // 128 x 129
  __shared__ float cA[64], sA[64];
  __shared__ int rnk[128];
  const int tid = threadIdx.x;
  const int bs  = blockDim.x;
  const int mat = blockIdx.x;

  for (int i = tid; i < 128 * 128; i += bs) {
    int r = i >> 7, c = i & 127;
    Am[r * 129 + c] = cov[(size_t)mat * (128 * 128) + i];
    Vm[r * 129 + c] = (r == c) ? 1.f : 0.f;
  }
  __syncthreads();

  for (int sweep = 0; sweep < 10; ++sweep) {
    for (int round = 0; round < 127; ++round) {
      if (tid < 64) {
        int p, q; pq_of(tid, round, &p, &q);
        float app = Am[p * 129 + p], aqq = Am[q * 129 + q], apq = Am[p * 129 + q];
        float c = 1.f, s = 0.f;
        if (fabsf(apq) > 1e-20f) {
          float tau = (aqq - app) / (2.f * apq);
          float t = 1.f / (fabsf(tau) + sqrtf(1.f + tau * tau));
          if (tau < 0.f) t = -t;
          c = 1.f / sqrtf(1.f + t * t);
          s = t * c;
        }
        cA[tid] = c; sA[tid] = s;
      }
      __syncthreads();
      // column rotations: A <- A*J and V <- V*J (16384 tasks: i fast)
      for (int t = tid; t < 16384; t += bs) {
        int i = t & 127, k = (t >> 7) & 63, w = t >> 13;
        int p, q; pq_of(k, round, &p, &q);
        float* Mx = w ? Vm : Am;
        float c = cA[k], s = sA[k];
        float x = Mx[i * 129 + p], y = Mx[i * 129 + q];
        Mx[i * 129 + p] = c * x - s * y;
        Mx[i * 129 + q] = s * x + c * y;
      }
      __syncthreads();
      // row rotations: A <- J^T * A (8192 tasks: j fast)
      for (int t = tid; t < 8192; t += bs) {
        int j = t & 127, k = t >> 7;
        int p, q; pq_of(k, round, &p, &q);
        float c = cA[k], s = sA[k];
        float x = Am[p * 129 + j], y = Am[q * 129 + j];
        Am[p * 129 + j] = c * x - s * y;
        Am[q * 129 + j] = s * x + c * y;
      }
      __syncthreads();
    }
  }

  // rank eigenvalues (diag) descending; copy top-16 eigenvector columns
  if (tid < 128) {
    float d = Am[tid * 129 + tid];
    int r = 0;
    for (int j = 0; j < 128; ++j) {
      float dj = Am[j * 129 + j];
      if (dj > d || (dj == d && j < tid)) r++;
    }
    rnk[tid] = r;
  }
  __syncthreads();
  for (int t = tid; t < 128 * 128; t += bs) {
    int q = t & 127, i = t >> 7;
    int r = rnk[q];
    if (r < KR) U[(size_t)mat * (128 * KR) + i * KR + r] = Vm[i * 129 + q];
  }
}

// =====================================================================
// K5: pair grams ||U_a^T U_b||_F^2  (WMMA f16 16x16x32, one wave per pair)
// blocks 0..143: teacher x teacher cross_gram (diag zeroed)
// blocks 144..191: student x teacher gram_sq
// =====================================================================
__global__ void pair_gram_kernel(const float* __restrict__ U,
                                 float* __restrict__ cg,
                                 float* __restrict__ gs) {
  const int b = blockIdx.x;
  const int lane = threadIdx.x;
  int ai, bi; float* outp; bool diag = false;
  if (b < 144) { ai = b / 12; bi = b % 12; outp = &cg[b]; diag = (ai == bi); }
  else { int t = b - 144; ai = 12 + t / 12; bi = t % 12; outp = &gs[t]; }
  const float* Ua = U + (size_t)ai * (128 * KR);
  const float* Ub = U + (size_t)bi * (128 * KR);
  const bool hi = lane >= 16;
  const int ii = lane & 15;

  v8f acc = {0.f,0.f,0.f,0.f,0.f,0.f,0.f,0.f};
  for (int kk = 0; kk < 128; kk += 32) {
    v16h a, bf;
    for (int j = 0; j < 16; ++j) {
      int k = kk + kmap16(j, hi);
      a[j]  = (_Float16)Ua[k * KR + ii];   // A = Ua^T : A[i][k] = Ua[k][i]
      bf[j] = (_Float16)Ub[k * KR + ii];   // B = Ub   : B[k][n] = Ub[k][n]
    }
    acc = wmma_f16_k32(a, bf, acc);
  }
  float s = 0.f;
  for (int v = 0; v < 8; ++v) s += acc[v] * acc[v];
  for (int off = 16; off > 0; off >>= 1) s += __shfl_xor(s, off, 32);
  if (lane == 0) *outp = diag ? 0.f : s;
}

// =====================================================================
// K6: softmax weights + diversity + reconstruction -> reg_loss
// single block of 64 threads
// =====================================================================
__global__ void weights_kernel(const float* __restrict__ gs,
                               const float* __restrict__ cg,
                               const float* __restrict__ log_temps,
                               const float* __restrict__ meanArr,
                               float* __restrict__ wts,
                               float* __restrict__ lossOut) {
  __shared__ float wL[PN * LN];
  __shared__ float red[64];
  __shared__ float orthSh;
  const int tid = threadIdx.x;
  if (tid == 0) {
    for (int p = 0; p < PN; ++p) {
      float tau = log1pf(expf(log_temps[p]));  // softplus
      float a[LN]; float mx = -1e30f;
      for (int l = 0; l < LN; ++l) {
        float dn = ((float)KR - gs[p * LN + l]) / (float)KR;
        a[l] = -dn / tau;
        mx = fmaxf(mx, a[l]);
      }
      float s = 0.f;
      for (int l = 0; l < LN; ++l) { a[l] = expf(a[l] - mx); s += a[l]; }
      for (int l = 0; l < LN; ++l) {
        float w = a[l] / s;
        wL[p * LN + l] = w;
        wts[p * LN + l] = w;
      }
    }
    float orth = 0.f;
    for (int p = 0; p < PN; ++p)
      for (int x = 0; x < LN; ++x)
        for (int y = 0; y < LN; ++y)
          orth += wL[p * LN + x] * wL[p * LN + y] * cg[x * LN + y];
    orthSh = 0.5f * orth;
  }
  __syncthreads();
  float acc = 0.f;
  for (int t = tid; t < PN * PD; t += 64) {
    int p = t >> 7, q = t & 127;
    float zm = 0.f;
    for (int l = 0; l < LN; ++l) zm += wL[p * LN + l] * meanArr[l * 128 + q];
    float d = zm - meanArr[(LN + p) * 128 + q];
    acc += d * d;
  }
  red[tid] = acc;
  __syncthreads();
  if (tid == 0) {
    float s = 0.f;
    for (int i = 0; i < 64; ++i) s += red[i];
    float recon = s / (float)PD;
    lossOut[0] = 0.01f * orthSh / (float)PN + 0.01f * recon / (float)PN;
  }
}

// =====================================================================
// K7/K8: streaming mix  out[p][x] = sum_l w[p][l]*src[l][x]   (float4, NT)
// =====================================================================
__global__ void mix_kernel(const float* __restrict__ src,
                           const float* __restrict__ wts,
                           float* __restrict__ dst, long n4) {
  __shared__ float w[PN * LN];
  if (threadIdx.x < PN * LN) w[threadIdx.x] = wts[threadIdx.x];
  __syncthreads();
  long x = (long)blockIdx.x * blockDim.x + threadIdx.x;
  if (x >= n4) return;
  const v4f* S = (const v4f*)src;
  v4f a0 = {0.f,0.f,0.f,0.f}, a1 = a0, a2 = a0, a3 = a0;
  for (int l = 0; l < LN; ++l) {
    v4f v = __builtin_nontemporal_load(&S[(size_t)l * n4 + x]);
    a0 += w[0 * LN + l] * v;
    a1 += w[1 * LN + l] * v;
    a2 += w[2 * LN + l] * v;
    a3 += w[3 * LN + l] * v;
  }
  v4f* D = (v4f*)dst;
  __builtin_nontemporal_store(a0, &D[x]);
  __builtin_nontemporal_store(a1, &D[(size_t)n4 + x]);
  __builtin_nontemporal_store(a2, &D[2 * (size_t)n4 + x]);
  __builtin_nontemporal_store(a3, &D[3 * (size_t)n4 + x]);
}

// =====================================================================
// host-side launch
// =====================================================================
extern "C" void kernel_launch(void* const* d_in, const int* in_sizes, int n_in,
                              void* d_out, int out_size, void* d_ws, size_t ws_size,
                              hipStream_t stream) {
  (void)in_sizes; (void)n_in; (void)out_size; (void)ws_size;
  const float* student = (const float*)d_in[0];  // [P,B,N,Ds]
  const float* teacher = (const float*)d_in[1];  // [L,B,N,Dt]
  const float* attns   = (const float*)d_in[2];  // [L,B,H,N,N]
  const float* phi_s   = (const float*)d_in[3];  // [128,Ds]
  const float* phi_t   = (const float*)d_in[4];  // [128,Dt]
  const float* ltemps  = (const float*)d_in[5];  // [P]
  float* out = (float*)d_out;

  // workspace layout (floats)
  float* ws   = (float*)d_ws;
  float* zAll = ws;                                  // 16 * MM * 128 (teachers 0..11, students 12..15)
  float* zt   = zAll;
  float* zs   = zAll + (size_t)LN * MM * PD;
  float* covW = zAll + (size_t)(LN + PN) * MM * PD;  // 16*128*128
  float* Uw   = covW + (size_t)16 * 128 * 128;       // 16*128*16
  float* mean = Uw + (size_t)16 * 128 * KR;          // 16*128
  float* cgW  = mean + 16 * 128;                     // 144
  float* gsW  = cgW + 144;                           // 48
  float* wtsW = gsW + 48;                            // 48

  // 1-2: projections (WMMA f32 16x16x4)
  proj_kernel<<<dim3(MM / 16, LN), 256, 0, stream>>>(teacher, phi_t, zt, DT);
  proj_kernel<<<dim3(MM / 16, PN), 256, 0, stream>>>(student, phi_s, zs, DS);
  // 3: means + covariances (WMMA f32 16x16x4)
  cov_kernel<<<16, 256, 0, stream>>>(zAll, covW, mean);
  // 4: Jacobi eigensolve + top-16 subspace
  jacobi_kernel<<<16, 512, 2 * 128 * 129 * sizeof(float), stream>>>(covW, Uw);
  // 5: pairwise subspace grams (WMMA f16 16x16x32)
  pair_gram_kernel<<<192, 32, 0, stream>>>(Uw, cgW, gsW);
  // 6: weights + reg_loss
  float* lossOut = out + PN * MTX + PN * MAX_;
  weights_kernel<<<1, 64, 0, stream>>>(gsW, cgW, ltemps, mean, wtsW, lossOut);
  // 7: mixed_teachers  (bandwidth bound: 232MB rd + 77MB wr)
  long n4t = (long)(MTX / 4);
  mix_kernel<<<(int)((n4t + 255) / 256), 256, 0, stream>>>(teacher, wtsW, out, n4t);
  // 8: mixed_attns  (bandwidth bound: 715MB rd + 238MB wr -> ~41us roofline)
  long n4a = (long)(MAX_ / 4);
  mix_kernel<<<(int)((n4a + 255) / 256), 256, 0, stream>>>(attns, wtsW, out + PN * MTX, n4a);
}
